// MambaSSM_5291399709428
// MI455X (gfx1250) — matrix operand order repaired and checked by reading, not compiled, not run on hardware
//
#include <hip/hip_runtime.h>
#include <hip/hip_bf16.h>

// ---------------------------------------------------------------------------
// Problem sizes (fixed by the reference)
// ---------------------------------------------------------------------------
#define TSEQ  32768
#define OBS   256
#define HID   512
#define NOBJ  4
#define NCOLGRP 4          // x_state, B, C, delta
#define SEGLEN 128
#define NSEG  (TSEQ / SEGLEN)   // 256

typedef __attribute__((ext_vector_type(16))) __bf16 v16bf;
typedef __attribute__((ext_vector_type(8)))  float  v8f;

union BfVec { v16bf v; uint4 q[2]; };

static __device__ __forceinline__ unsigned short f32_to_bf16(float f) {
    unsigned int u = __float_as_uint(f);
    u += 0x7FFFu + ((u >> 16) & 1u);     // round to nearest even
    return (unsigned short)(u >> 16);
}

// ---------------------------------------------------------------------------
// 1. f32 -> bf16 row-major conversion (activations)
// ---------------------------------------------------------------------------
__global__ void mamba_cvt_bf16(const float* __restrict__ src,
                               unsigned short* __restrict__ dst, int n) {
    int i = blockIdx.x * blockDim.x + threadIdx.x;
    if (i < n) dst[i] = f32_to_bf16(src[i]);
}

// ---------------------------------------------------------------------------
// 2. Pack one weight matrix W[K,HID] (f32 row-major) into the CDNA5 WMMA
//    bf16 B-operand layout. Packed element layout per 32x16 (KxN) tile:
//      lane = (n%16) + 16*((k_rel>>3)&1), within-lane bf16 index = (k_rel&7) + 8*(k_rel>=16)
//    flat:  (((g*32 + n/16)*KT + k/32)*32 + lane)*16 + ((k_rel&7) + 8*(k_rel>=16))
//    so each lane's 16 bf16 (32B) for a tile are contiguous.
// ---------------------------------------------------------------------------
__global__ void mamba_pack_w(const float* __restrict__ W,
                             unsigned short* __restrict__ dst,
                             int K, int group) {
    int idx = blockIdx.x * blockDim.x + threadIdx.x;   // k*HID + c
    if (idx >= K * HID) return;
    int k = idx >> 9;            // /512
    int c = idx & 511;
    int kt = k >> 5;
    int kr = k & 31;
    int KT = K >> 5;
    int ntile = group * 32 + (c >> 4);
    int lane  = (c & 15) + ((kr & 8) ? 16 : 0);
    size_t off = (((size_t)(ntile * KT + kt)) * 32 + lane) * 16 + (size_t)((kr & 7) | ((kr & 16) >> 1));
    dst[off] = f32_to_bf16(W[idx]);
}

// ---------------------------------------------------------------------------
// 3. Fused 4-projection GEMM + SSM pointwise epilogue.
//    One wave -> one 16x16 (t,h) tile with 4 f32 accumulators.
//    Emits A_bar, Bx, C directly (the [T,2048] projection never hits memory).
// ---------------------------------------------------------------------------
__global__ __launch_bounds__(256)
void mamba_gemm4(const unsigned short* __restrict__ Abf,   // [TSEQ x K] bf16
                 const unsigned short* __restrict__ Wpk,   // packed weights
                 const float* __restrict__ A_log,          // [HID]
                 float* __restrict__ Abar,                  // [TSEQ x HID]
                 float* __restrict__ Bx,                    // [TSEQ x HID]
                 float* __restrict__ Cout,                  // [TSEQ x HID]
                 int K) {
    const int gtid  = blockIdx.x * blockDim.x + threadIdx.x;
    const int wave  = gtid >> 5;
    const int lane  = gtid & 31;
    const int htile = wave & 31;          // HID/16 = 32 tiles
    const int mtile = wave >> 5;          // TSEQ/16 = 2048 tiles
    const int KT    = K >> 5;

    // A operand (16-bit A 16x32 layout): lanes 0-15 take K=[0..7]+[16..23],
    // lanes 16-31 take K=[8..15]+[24..31] -> two contiguous 16B chunks.
    const int arow  = mtile * 16 + (lane & 15);
    const int abase = (lane & 16) ? 8 : 0;
    const unsigned short* Ap = Abf + (size_t)arow * K + abase;

    v8f acc[NCOLGRP] = {v8f{}, v8f{}, v8f{}, v8f{}};

    for (int kt = 0; kt < KT; ++kt) {
        BfVec a;
        a.q[0] = *(const uint4*)(Ap + kt * 32);
        a.q[1] = *(const uint4*)(Ap + kt * 32 + 16);
        if (kt + 1 < KT) __builtin_prefetch(Ap + (kt + 1) * 32, 0, 0);
#pragma unroll
        for (int g = 0; g < NCOLGRP; ++g) {
            const uint4* bp = (const uint4*)(Wpk +
                (((size_t)((g * 32 + htile) * KT + kt)) * 32 + lane) * 16);
            BfVec b;
            b.q[0] = bp[0];
            b.q[1] = bp[1];
            acc[g] = __builtin_amdgcn_wmma_f32_16x16x32_bf16(
                false, a.v, false, b.v, (short)0, acc[g], false, false);
        }
    }

    // D layout: VGPR j / lane l -> row M = j + 8*(l>=16), col N = l%16
    const int hcol  = htile * 16 + (lane & 15);
    const float negA = -__expf(A_log[hcol]);
    const int trow0 = mtile * 16 + ((lane & 16) ? 8 : 0);
#pragma unroll
    for (int j = 0; j < 8; ++j) {
        size_t idx = (size_t)(trow0 + j) * HID + hcol;
        float xs = acc[0][j];
        float bb = acc[1][j];
        float cc = acc[2][j];
        float dd = acc[3][j];
        float delta = 1.0f / (1.0f + __expf(-dd));
        Abar[idx] = __expf(delta * negA);
        Bx[idx]   = bb * xs;
        Cout[idx] = cc;
    }
}

// ---------------------------------------------------------------------------
// 4. Segmented scan for h_t = a_t*h_{t-1} + b_t   (3 passes)
// ---------------------------------------------------------------------------
__global__ void mamba_scan1(const float* __restrict__ Abar,
                            const float* __restrict__ Bx,
                            float* __restrict__ Pseg,
                            float* __restrict__ Sseg) {
    int gid = blockIdx.x * blockDim.x + threadIdx.x;   // seg*HID + h
    if (gid >= NSEG * HID) return;
    int h   = gid & (HID - 1);
    int seg = gid >> 9;
    size_t base = (size_t)seg * SEGLEN * HID + h;
    float p = 1.0f, s = 0.0f;
    for (int i = 0; i < SEGLEN; ++i) {
        float a = Abar[base + (size_t)i * HID];
        float b = Bx  [base + (size_t)i * HID];
        s = a * s + b;
        p *= a;
    }
    Pseg[gid] = p;
    Sseg[gid] = s;
}

__global__ void mamba_scan2(const float* __restrict__ Pseg,
                            const float* __restrict__ Sseg,
                            float* __restrict__ Carry) {
    int h = blockIdx.x * blockDim.x + threadIdx.x;
    if (h >= HID) return;
    float c = 0.0f;
    for (int seg = 0; seg < NSEG; ++seg) {
        Carry[(size_t)seg * HID + h] = c;
        c = Pseg[(size_t)seg * HID + h] * c + Sseg[(size_t)seg * HID + h];
    }
}

// pass 3: apply carries, fuse y = C*h; optionally emit bf16 (next layer input)
// or overwrite Cio in place with y (f32, for the output head).
__global__ void mamba_scan3(const float* __restrict__ Abar,
                            const float* __restrict__ Bx,
                            const float* __restrict__ Carry,
                            float* __restrict__ Cio,
                            unsigned short* __restrict__ ybf /*may be null*/,
                            int emit_bf16) {
    int gid = blockIdx.x * blockDim.x + threadIdx.x;
    if (gid >= NSEG * HID) return;
    int h   = gid & (HID - 1);
    int seg = gid >> 9;
    size_t base = (size_t)seg * SEGLEN * HID + h;
    float hh = Carry[gid];
    for (int i = 0; i < SEGLEN; ++i) {
        size_t idx = base + (size_t)i * HID;
        float a = Abar[idx];
        float b = Bx[idx];
        hh = a * hh + b;
        float y = Cio[idx] * hh;
        if (emit_bf16) ybf[idx] = f32_to_bf16(y);
        else           Cio[idx] = y;
    }
}

// ---------------------------------------------------------------------------
// 5. Output head: out = y1 @ W_out + b_out + x @ W_skip   (N = 4)
// ---------------------------------------------------------------------------
__global__ void mamba_head(const float* __restrict__ y1,    // [TSEQ x HID]
                           const float* __restrict__ x,     // [TSEQ x OBS]
                           const float* __restrict__ Wout,  // [HID x 4]
                           const float* __restrict__ bout,  // [4]
                           const float* __restrict__ Wskip, // [OBS x 4]
                           float* __restrict__ out) {       // [TSEQ x 4]
    int t = blockIdx.x * blockDim.x + threadIdx.x;
    if (t >= TSEQ) return;
    float a0 = bout[0], a1 = bout[1], a2 = bout[2], a3 = bout[3];
    const float* yr = y1 + (size_t)t * HID;
    for (int h = 0; h < HID; ++h) {
        float yv = yr[h];
        float4 w = ((const float4*)Wout)[h];
        a0 += yv * w.x; a1 += yv * w.y; a2 += yv * w.z; a3 += yv * w.w;
    }
    const float* xr = x + (size_t)t * OBS;
    for (int d = 0; d < OBS; ++d) {
        float xv = xr[d];
        float4 w = ((const float4*)Wskip)[d];
        a0 += xv * w.x; a1 += xv * w.y; a2 += xv * w.z; a3 += xv * w.w;
    }
    ((float4*)out)[t] = make_float4(a0, a1, a2, a3);
}

// ---------------------------------------------------------------------------
// Launch
// ---------------------------------------------------------------------------
extern "C" void kernel_launch(void* const* d_in, const int* in_sizes, int n_in,
                              void* d_out, int out_size, void* d_ws, size_t ws_size,
                              hipStream_t stream) {
    const float* x      = (const float*)d_in[0];
    const float* W0[4]  = {(const float*)d_in[1], (const float*)d_in[2],
                           (const float*)d_in[3], (const float*)d_in[4]};
    const float* A_log0 = (const float*)d_in[5];
    const float* W1[4]  = {(const float*)d_in[6], (const float*)d_in[7],
                           (const float*)d_in[8], (const float*)d_in[9]};
    const float* A_log1 = (const float*)d_in[10];
    const float* W_out  = (const float*)d_in[11];
    const float* b_out  = (const float*)d_in[12];
    const float* W_skip = (const float*)d_in[13];
    float* out = (float*)d_out;

    // workspace carve-up (256B aligned)
    char* ws = (char*)d_ws;
    size_t off = 0;
    auto carve = [&](size_t bytes) -> char* {
        char* p = ws + off;
        off = (off + bytes + 255) & ~(size_t)255;
        return p;
    };
    unsigned short* x_bf   = (unsigned short*)carve((size_t)TSEQ * OBS * 2);
    unsigned short* y0_bf  = (unsigned short*)carve((size_t)TSEQ * HID * 2);
    float* Abar            = (float*)carve((size_t)TSEQ * HID * 4);
    float* Bx              = (float*)carve((size_t)TSEQ * HID * 4);
    float* Cbuf            = (float*)carve((size_t)TSEQ * HID * 4);
    unsigned short* pW0    = (unsigned short*)carve((size_t)OBS * NCOLGRP * HID * 2);
    unsigned short* pW1    = (unsigned short*)carve((size_t)HID * NCOLGRP * HID * 2);
    float* Pseg            = (float*)carve((size_t)NSEG * HID * 4);
    float* Sseg            = (float*)carve((size_t)NSEG * HID * 4);
    float* Carry           = (float*)carve((size_t)NSEG * HID * 4);
    (void)ws_size; (void)n_in; (void)in_sizes; (void)out_size;

    // 1. activations -> bf16
    {
        int n = TSEQ * OBS;
        mamba_cvt_bf16<<<(n + 255) / 256, 256, 0, stream>>>(x, x_bf, n);
    }
    // 2. pack weights (layer 0: K=OBS, layer 1: K=HID)
    for (int g = 0; g < 4; ++g) {
        int n0 = OBS * HID;
        mamba_pack_w<<<(n0 + 255) / 256, 256, 0, stream>>>(W0[g], pW0, OBS, g);
        int n1 = HID * HID;
        mamba_pack_w<<<(n1 + 255) / 256, 256, 0, stream>>>(W1[g], pW1, HID, g);
    }

    const int gemm_waves  = (TSEQ / 16) * (HID / 16);      // 65536
    const int gemm_blocks = gemm_waves * 32 / 256;         // 8192
    const int scan_thr    = NSEG * HID;                    // 131072

    // 3. layer 0: fused projections + SSM pointwise
    mamba_gemm4<<<gemm_blocks, 256, 0, stream>>>(x_bf, pW0, A_log0,
                                                 Abar, Bx, Cbuf, OBS);
    // 4. layer 0 scan -> y0 (bf16)
    mamba_scan1<<<scan_thr / 256, 256, 0, stream>>>(Abar, Bx, Pseg, Sseg);
    mamba_scan2<<<HID / 256, 256, 0, stream>>>(Pseg, Sseg, Carry);
    mamba_scan3<<<scan_thr / 256, 256, 0, stream>>>(Abar, Bx, Carry, Cbuf,
                                                    y0_bf, 1);

    // 5. layer 1: fused projections + SSM pointwise
    mamba_gemm4<<<gemm_blocks, 256, 0, stream>>>(y0_bf, pW1, A_log1,
                                                 Abar, Bx, Cbuf, HID);
    // 6. layer 1 scan -> y1 (f32, in place over Cbuf)
    mamba_scan1<<<scan_thr / 256, 256, 0, stream>>>(Abar, Bx, Pseg, Sseg);
    mamba_scan2<<<HID / 256, 256, 0, stream>>>(Pseg, Sseg, Carry);
    mamba_scan3<<<scan_thr / 256, 256, 0, stream>>>(Abar, Bx, Carry, Cbuf,
                                                    (unsigned short*)nullptr, 0);

    // 7. output head
    mamba_head<<<(TSEQ + 255) / 256, 256, 0, stream>>>(Cbuf, x, W_out, b_out,
                                                       W_skip, out);
}
